// FIRE_35631048688287
// MI455X (gfx1250) — matrix operand order repaired and checked by reading, not compiled, not run on hardware
//
#include <hip/hip_runtime.h>
#include <hip/hip_bf16.h>

typedef __attribute__((ext_vector_type(16))) _Float16 v16h;
typedef __attribute__((ext_vector_type(8)))  float    v8f;

#define FIRE_S 2048
#define FIRE_H 12
#define FIRE_W 32
#define FIRE_S2 (FIRE_S * FIRE_S)          // 4194304, fits int
#define FIRE_EPS 1e-6f
#define FIRE_LOG_BIAS 1.0f

// ---------------------------------------------------------------------------
// Kernel 1: collapse the 4.2M logs to 2*S table entries.
//   tab[d]     = log((|d|+eps)*c + 1 + eps)                (log_rel, d = |i-j|)
//   tab[S + i] = 1 / log(|c * (max(i,thr)+eps)| + 1 + eps) (1/log_norm)
// ---------------------------------------------------------------------------
__global__ void fire_tables_kernel(const float* __restrict__ c_p,
                                   const float* __restrict__ Lm_p,
                                   const float* __restrict__ iL_p,
                                   float* __restrict__ tab) {
    int t = blockIdx.x * blockDim.x + threadIdx.x;
    if (t >= FIRE_S) return;
    float c   = c_p[0];
    float thr = fabsf(Lm_p[0] * iL_p[0]);
    float d   = (float)t;
    tab[t] = logf((d + FIRE_EPS) * c + FIRE_LOG_BIAS + FIRE_EPS);
    float pn = fmaxf(d, thr) + FIRE_EPS;
    tab[FIRE_S + t] = 1.0f / logf(fabsf(c * pn) + FIRE_LOG_BIAS + FIRE_EPS);
}

// ---------------------------------------------------------------------------
// Kernel 2: wave builds B = 32x16 f16 activation tile (16 (i,j) pairs),
// A = w2 padded 16x32 f16 (loaded once per wave), C init = b2 per head,
// one v_wmma_f32_16x16x32_f16 per tile, NT stores of 12 heads x 16 floats.
//
// Store addressing: single uniform SGPR base (out); the per-head stride
// (v*S2 elements) and lane-half stride (8*S2) live in 8 precomputed 32-bit
// per-lane voffset registers (max ~50M elements < 2^31). The only per-
// iteration term is it*128 elements = it*512 bytes, which folds into the
// store's signed 24-bit IOFFSET immediate -> no address VALU in the loop.
//
// Block = 256 threads = 8 waves; each wave does 4 tiles -> block covers 512 j.
// Grid = (S/512, S).
// ---------------------------------------------------------------------------
__global__ void __launch_bounds__(256)
fire_bias_kernel(const float* __restrict__ w1, const float* __restrict__ b1,
                 const float* __restrict__ w2, const float* __restrict__ b2,
                 const float* __restrict__ tab, float* __restrict__ out) {
    const int lane    = threadIdx.x & 31;
    const int wave    = threadIdx.x >> 5;
    const int i       = blockIdx.y;
    const int jbase   = blockIdx.x * 512 + wave * 16;   // this wave's first j
    const int col     = lane & 15;                      // pair column / A row
    const bool hiHalf = (lane >= 16);

    // --- A-matrix: w2 (H x W) zero-padded to 16 x 32, contiguous K order (k = 16*half + e), same as B.
    // lane<16 : M=lane,    K = {0..15}
    // lane>=16: M=lane-16, K = {16..31}
    v16h A;
    if (col < FIRE_H) {
        const int kb0 = hiHalf ? 16 : 0;
        const int kb1 = hiHalf ? 24 : 8;
        #pragma unroll
        for (int e = 0; e < 8; ++e) {
            A[e]     = (_Float16)w2[col * FIRE_W + kb0 + e];
            A[e + 8] = (_Float16)w2[col * FIRE_W + kb1 + e];
        }
    } else {
        #pragma unroll
        for (int e = 0; e < 16; ++e) A[e] = (_Float16)0.0f;
    }

    // --- Per-lane layer-1 params in f16. B-matrix lane holds K = wbase + e.
    const int wbase = hiHalf ? 16 : 0;
    v16h w1v, b1v;
    #pragma unroll
    for (int e = 0; e < 16; ++e) {
        w1v[e] = (_Float16)w1[wbase + e];
        b1v[e] = (_Float16)b1[wbase + e];
    }

    // --- Bias per C register: C VGPR v -> head v (lanes 0-15) / v+8 (16-31).
    float b2r[8];
    #pragma unroll
    for (int v = 0; v < 8; ++v) {
        const int head = v + (hiHalf ? 8 : 0);
        b2r[v] = (head < FIRE_H) ? b2[head] : 0.0f;
    }

    const float iln = tab[FIRE_S + i];

    // --- Per-lane 32-bit store offsets (elements), one per C VGPR,
    // computed once. Hi half adds 8 head strides. For v>=4 only lanes<16
    // (hioff==0) ever store, so the hioff term there is harmless.
    const int hioff = hiHalf ? 8 * FIRE_S2 : 0;
    const int off0  = i * FIRE_S + jbase + col;
    int voff[8];
    #pragma unroll
    for (int v = 0; v < 8; ++v) voff[v] = v * FIRE_S2 + hioff + off0;

    #pragma unroll
    for (int it = 0; it < 4; ++it) {
        const int j = jbase + it * 128 + col;
        int d = i - j; d = (d < 0) ? -d : d;
        const _Float16 ndh = (_Float16)(tab[d] * iln);

        // B-matrix: h[K] = relu(nd * w1[K] + b1[K]), f16 throughout
        // (-> v_pk_fma_f16 / v_pk_max_num_f16).
        v16h B;
        #pragma unroll
        for (int e = 0; e < 16; ++e) {
            const _Float16 h = (_Float16)(ndh * w1v[e] + b1v[e]);
            B[e] = (h > (_Float16)0.0f) ? h : (_Float16)0.0f;
        }

        // C init = b2 (bias folded into accumulator).
        v8f C;
        #pragma unroll
        for (int v = 0; v < 8; ++v) C[v] = b2r[v];

        C = __builtin_amdgcn_wmma_f32_16x16x32_f16(
                /*neg_a=*/false, A, /*neg_b=*/false, B,
                /*c_mod=*/(short)0, C, /*reuse_a=*/false, /*reuse_b=*/false);

        // Heads 0..3 and 8..11 via v=0..3: unconditional full-wave stores;
        // it*128 elements folds into the IOFFSET immediate.
        #pragma unroll
        for (int v = 0; v < 4; ++v) {
            __builtin_nontemporal_store(C[v], out + voff[v] + it * 128);
        }
        // Heads 4..7 live in lanes 0-15 of v=4..7; single divergence region.
        if (!hiHalf) {
            #pragma unroll
            for (int v = 4; v < 8; ++v) {
                __builtin_nontemporal_store(C[v], out + voff[v] + it * 128);
            }
        }
    }
}

extern "C" void kernel_launch(void* const* d_in, const int* in_sizes, int n_in,
                              void* d_out, int out_size, void* d_ws, size_t ws_size,
                              hipStream_t stream) {
    // inputs: x(unused), w1, b1, w2, b2, c, L_multiplier, init_L
    const float* w1  = (const float*)d_in[1];
    const float* b1  = (const float*)d_in[2];
    const float* w2  = (const float*)d_in[3];
    const float* b2  = (const float*)d_in[4];
    const float* c   = (const float*)d_in[5];
    const float* Lm  = (const float*)d_in[6];
    const float* iL  = (const float*)d_in[7];
    float* out = (float*)d_out;
    float* tab = (float*)d_ws;          // 2*S floats = 16 KB

    fire_tables_kernel<<<dim3(FIRE_S / 256), dim3(256), 0, stream>>>(c, Lm, iL, tab);

    fire_bias_kernel<<<dim3(FIRE_S / 512, FIRE_S), dim3(256), 0, stream>>>(
        w1, b1, w2, b2, tab, out);
}
